// SpectralLoss_1417339207876
// MI455X (gfx1250) — matrix operand order repaired and checked by reading, not compiled, not run on hardware
//
#include <hip/hip_runtime.h>
#include <hip/hip_bf16.h>
#include <math.h>

typedef __attribute__((ext_vector_type(16))) _Float16 v16h;
typedef __attribute__((ext_vector_type(8)))  float    v8f;
typedef __attribute__((ext_vector_type(4)))  float    v4f;

#define HH   256        // image rows
#define WW   256        // image cols
#define WC   129        // rfft one-sided cols
#define NPAD 144        // 9 n-tiles of 16
#define NT9  9
#define BIMG 128

// ---- workspace layout (bytes) ----
// Stage-1 B twiddles (rDFT, pre-swizzled B-operand layout, hi/lo f16): [8 kt][9 nt][32 lane][16 h]
static constexpr size_t OFF_CRH  = 0;
static constexpr size_t OFF_CRL  = 73728;
static constexpr size_t OFF_CIH  = 147456;
static constexpr size_t OFF_CIL  = 221184;
// Stage-2 A twiddles (DFT, pre-swizzled A-operand layout, hi/lo f16): [16 mt][8 kt][32 lane][16 h]
static constexpr size_t OFF_DRH  = 294912;
static constexpr size_t OFF_DRL  = 425984;
static constexpr size_t OFF_DIH  = 557056;
static constexpr size_t OFF_DIL  = 688128;
static constexpr size_t OFF_DNH  = 819200;   // -Im(D) hi
static constexpr size_t OFF_DNL  = 950272;   // -Im(D) lo
// fft_abs f32 [128][256][144]
static constexpr size_t OFF_FAB  = 1081344;
// Stage-1 output G1 (pre-swizzled stage-2 B-operand layout, hi/lo f16): [128][8 kbt][9 nt][32][16]
static constexpr size_t OFF_G1RH = 19955712;
static constexpr size_t OFF_G1RL = 29392896;
static constexpr size_t OFF_G1IH = 38830080;
static constexpr size_t OFF_G1IL = 48267264;
static constexpr size_t OFF_PART = 57704448; // 128 f32 partial losses

__device__ __forceinline__ v8f wmma16(v16h a, v16h b, v8f c) {
  // D = A(16x32 f16) * B(32x16 f16) + C(16x16 f32)
  return __builtin_amdgcn_wmma_f32_16x16x32_f16(false, a, false, b, (short)0, c, false, false);
}

__device__ __forceinline__ void split2(float x, _Float16& hi, _Float16& lo) {
  _Float16 h = (_Float16)x;
  hi = h;
  lo = (_Float16)(x - (float)h);
}

// ---------- twiddle init: stage-1 rDFT matrix C[w][n] = exp(-2*pi*i*w*n/256), B-operand swizzle ----------
__global__ void k_init_stage1B(char* ws) {
  int e = blockIdx.x * blockDim.x + threadIdx.x;
  if (e >= 8 * NT9 * 32 * 16) return;
  int j    = e & 15;
  int lane = (e >> 4) & 31;
  int t    = e >> 9;            // 0..71
  int nt   = t % NT9;
  int kt   = t / NT9;
  // B layout (32x16): lanes 0-15 -> N=lane, K = {0..7, 16..23} ; lanes 16-31 -> N=lane-16, K = {8..15, 24..31}
  int n = nt * 16 + (lane & 15);
  int k = kt * 32 + ((lane >> 4) ? 8 : 0) + ((j < 8) ? j : (j + 8));
  float cr = 0.f, ci = 0.f;
  if (n < WC) {
    int m = (k * n) & 255;                       // phase periodicity keeps angle exact
    float th = 6.28318530717958647692f * (float)m * (1.0f / 256.0f);
    float s, c;
    sincosf(th, &s, &c);
    cr = c;          // Re e^{-i th}
    ci = -s;         // Im e^{-i th}
  }
  _Float16 h, l;
  _Float16* Crh = (_Float16*)(ws + OFF_CRH);
  _Float16* Crl = (_Float16*)(ws + OFF_CRL);
  _Float16* Cih = (_Float16*)(ws + OFF_CIH);
  _Float16* Cil = (_Float16*)(ws + OFF_CIL);
  split2(cr, h, l); Crh[e] = h; Crl[e] = l;
  split2(ci, h, l); Cih[e] = h; Cil[e] = l;
}

// ---------- twiddle init: stage-2 DFT matrix D[r][h] = exp(-2*pi*i*r*h/256), A-operand swizzle ----------
__global__ void k_init_stage2A(char* ws) {
  int e = blockIdx.x * blockDim.x + threadIdx.x;
  if (e >= 16 * 8 * 32 * 16) return;
  int j    = e & 15;
  int lane = (e >> 4) & 31;
  int kt   = (e >> 9) & 7;
  int mt   = e >> 12;
  // A layout (16x32): lanes 0-15 -> M=lane, K = khalf + {0..7, 16..23}; lanes 16-31 -> M=lane-16, khalf=8
  int M     = mt * 16 + (lane & 15);
  int khalf = (lane >> 4) ? 8 : 0;
  int K     = kt * 32 + khalf + ((j < 8) ? j : (j + 8));
  int m = (M * K) & 255;
  float th = 6.28318530717958647692f * (float)m * (1.0f / 256.0f);
  float s, c;
  sincosf(th, &s, &c);
  _Float16 h, l;
  _Float16* Drh = (_Float16*)(ws + OFF_DRH);
  _Float16* Drl = (_Float16*)(ws + OFF_DRL);
  _Float16* Dih = (_Float16*)(ws + OFF_DIH);
  _Float16* Dil = (_Float16*)(ws + OFF_DIL);
  _Float16* Dnh = (_Float16*)(ws + OFF_DNH);
  _Float16* Dnl = (_Float16*)(ws + OFF_DNL);
  split2(c,  h, l); Drh[e] = h; Drl[e] = l;   // Re
  split2(-s, h, l); Dih[e] = h; Dil[e] = l;   // Im
  split2(s,  h, l); Dnh[e] = h; Dnl[e] = l;   // -Im  (Fr = Dr*G1r + (-Di)*G1i)
}

// ---------- stage 1: fused luma + G1 = gray @ C (rFFT along W), split-f16 3-product WMMA ----------
// One workgroup = 4 waves = 64 output rows, ALL 9 n-tiles (input rows read exactly once).
__global__ void __launch_bounds__(128) k_stage1(const float* __restrict__ data, char* ws) {
  const int lane = threadIdx.x & 31;
  const int wave = threadIdx.x >> 5;       // 0..3
  const int mblk = blockIdx.x;             // 0..3 (64 rows each)
  const int img  = blockIdx.y;             // 0..127

  const v16h* Crh = (const v16h*)(ws + OFF_CRH);
  const v16h* Crl = (const v16h*)(ws + OFF_CRL);
  const v16h* Cih = (const v16h*)(ws + OFF_CIH);
  const v16h* Cil = (const v16h*)(ws + OFF_CIL);

  const int mrow  = mblk * 64 + wave * 16 + (lane & 15);
  const int khalf = (lane >> 4) ? 8 : 0;

  const float* __restrict__ pR = data + (size_t)img * 3 * 65536 + (size_t)mrow * 256;
  const float* __restrict__ pG = pR + 65536;
  const float* __restrict__ pB = pR + 131072;

  v8f accR[NT9], accI[NT9];
  const v8f z = {0.f, 0.f, 0.f, 0.f, 0.f, 0.f, 0.f, 0.f};
#pragma unroll
  for (int nt = 0; nt < NT9; ++nt) { accR[nt] = z; accI[nt] = z; }

  for (int kt = 0; kt < 8; ++kt) {
    const int o = kt * 32 + khalf;
    v4f r0 = *(const v4f*)(pR + o),      r1 = *(const v4f*)(pR + o + 4);
    v4f r2 = *(const v4f*)(pR + o + 16), r3 = *(const v4f*)(pR + o + 20);
    v4f g0 = *(const v4f*)(pG + o),      g1 = *(const v4f*)(pG + o + 4);
    v4f g2 = *(const v4f*)(pG + o + 16), g3 = *(const v4f*)(pG + o + 20);
    v4f b0 = *(const v4f*)(pB + o),      b1 = *(const v4f*)(pB + o + 4);
    v4f b2 = *(const v4f*)(pB + o + 16), b3 = *(const v4f*)(pB + o + 20);
    if (kt < 7) {                        // CDNA5 global_prefetch_b8 for next K-step rows
      __builtin_prefetch(pR + o + 32);
      __builtin_prefetch(pG + o + 32);
      __builtin_prefetch(pB + o + 32);
    }
    v4f y0 = 0.299f * r0 + 0.587f * g0 + 0.114f * b0;
    v4f y1 = 0.299f * r1 + 0.587f * g1 + 0.114f * b1;
    v4f y2 = 0.299f * r2 + 0.587f * g2 + 0.114f * b2;
    v4f y3 = 0.299f * r3 + 0.587f * g3 + 0.114f * b3;
    v16h Ah, Al;
#pragma unroll
    for (int j = 0; j < 4; ++j) {
      _Float16 h, l;
      split2(y0[j], h, l); Ah[j]      = h; Al[j]      = l;
      split2(y1[j], h, l); Ah[4 + j]  = h; Al[4 + j]  = l;
      split2(y2[j], h, l); Ah[8 + j]  = h; Al[8 + j]  = l;
      split2(y3[j], h, l); Ah[12 + j] = h; Al[12 + j] = l;
    }
#pragma unroll
    for (int nt = 0; nt < NT9; ++nt) {
      int bidx = (kt * NT9 + nt) * 32 + lane;
      v16h brh = Crh[bidx], brl = Crl[bidx];
      v16h bih = Cih[bidx], bil = Cil[bidx];
      accR[nt] = wmma16(Ah, brh, accR[nt]);
      accR[nt] = wmma16(Ah, brl, accR[nt]);
      accR[nt] = wmma16(Al, brh, accR[nt]);
      accI[nt] = wmma16(Ah, bih, accI[nt]);
      accI[nt] = wmma16(Ah, bil, accI[nt]);
      accI[nt] = wmma16(Al, bih, accI[nt]);
    }
  }

  // Transpose D-layout (16x16 f32) -> stage-2 B-operand layout via LDS, split hi/lo f16, store.
  __shared__ float ldsT[2][64][16];
  const int ncol  = lane & 15;
  const int mofs  = (lane >> 4) ? 8 : 0;
  const int comp  = wave >> 1;                       // 0 -> G1r, 1 -> G1i
  const int bt    = wave & 1;                        // 32-row K group within this block
  const int rbase = bt * 32 + ((lane >> 4) ? 8 : 0);
  const int kbt   = mblk * 2 + bt;                   // global 32-row K tile (0..7)
  v16h* Gh = (v16h*)(ws + (comp ? OFF_G1IH : OFF_G1RH));
  v16h* Gl = (v16h*)(ws + (comp ? OFF_G1IL : OFF_G1RL));
#pragma unroll
  for (int nt = 0; nt < NT9; ++nt) {
    __syncthreads();
#pragma unroll
    for (int v = 0; v < 8; ++v) {
      ldsT[0][wave * 16 + mofs + v][ncol] = accR[nt][v];
      ldsT[1][wave * 16 + mofs + v][ncol] = accI[nt][v];
    }
    __syncthreads();
    v16h vh, vl;
#pragma unroll
    for (int j = 0; j < 16; ++j) {
      float x = ldsT[comp][rbase + ((j < 8) ? j : (j + 8))][ncol];
      _Float16 h, l;
      split2(x, h, l);
      vh[j] = h; vl[j] = l;
    }
    size_t o = (((size_t)img * 8 + kbt) * NT9 + nt) * 32 + lane;
    Gh[o] = vh;
    Gl[o] = vl;
  }
}

// ---------- stage 2: F = D @ G1 (FFT along H, complex), fused power + log ----------
// One workgroup = 16 waves = full M=256 for one n-tile; G1 read exactly once.
__global__ void __launch_bounds__(512) k_stage2(char* ws) {
  const int lane = threadIdx.x & 31;
  const int mt   = threadIdx.x >> 5;                 // wave id = output freq-row tile 0..15
  const int nt   = blockIdx.x;
  const int img  = blockIdx.y;

  const v16h* Drh = (const v16h*)(ws + OFF_DRH);
  const v16h* Drl = (const v16h*)(ws + OFF_DRL);
  const v16h* Dih = (const v16h*)(ws + OFF_DIH);
  const v16h* Dil = (const v16h*)(ws + OFF_DIL);
  const v16h* Dnh = (const v16h*)(ws + OFF_DNH);
  const v16h* Dnl = (const v16h*)(ws + OFF_DNL);
  const v16h* G1rh = (const v16h*)(ws + OFF_G1RH);
  const v16h* G1rl = (const v16h*)(ws + OFF_G1RL);
  const v16h* G1ih = (const v16h*)(ws + OFF_G1IH);
  const v16h* G1il = (const v16h*)(ws + OFF_G1IL);

  v8f Fr = {0.f, 0.f, 0.f, 0.f, 0.f, 0.f, 0.f, 0.f};
  v8f Fi = Fr;

  for (int kt = 0; kt < 8; ++kt) {
    size_t ai = ((size_t)mt * 8 + kt) * 32 + lane;
    v16h drh = Drh[ai], drl = Drl[ai];
    v16h dih = Dih[ai], dil = Dil[ai];
    v16h dnh = Dnh[ai], dnl = Dnl[ai];
    size_t bi = (((size_t)img * 8 + kt) * NT9 + nt) * 32 + lane;
    v16h brh = G1rh[bi], brl = G1rl[bi];
    v16h bih = G1ih[bi], bil = G1il[bi];
    // Fr = Dr*G1r - Di*G1i  (subtraction folded into pre-negated Dn = -Di)
    Fr = wmma16(drh, brh, Fr); Fr = wmma16(drh, brl, Fr); Fr = wmma16(drl, brh, Fr);
    Fr = wmma16(dnh, bih, Fr); Fr = wmma16(dnh, bil, Fr); Fr = wmma16(dnl, bih, Fr);
    // Fi = Dr*G1i + Di*G1r
    Fi = wmma16(drh, bih, Fi); Fi = wmma16(drh, bil, Fi); Fi = wmma16(drl, bih, Fi);
    Fi = wmma16(dih, brh, Fi); Fi = wmma16(dih, brl, Fi); Fi = wmma16(dil, brh, Fi);
  }

  // fused power spectrum + 20*ln(p + eps); row-major [256][144] (pad cols unread)
  float* fab = (float*)(ws + OFF_FAB) + (size_t)img * HH * NPAD;
  const int col = nt * 16 + (lane & 15);
  const int r0  = mt * 16 + ((lane >> 4) ? 8 : 0);
#pragma unroll
  for (int v = 0; v < 8; ++v) {
    float p = Fr[v] * Fr[v] + Fi[v] * Fi[v];
    fab[(size_t)(r0 + v) * NPAD + col] = 20.0f * logf(p + 1e-8f);
  }
}

// ---------- radial profile + min-max normalize + L1 vs mean (per image) ----------
__global__ void __launch_bounds__(256) k_profile(char* ws, const int* __restrict__ radius,
                                                 const float* __restrict__ mean, int L,
                                                 float* __restrict__ partials) {
  __shared__ float s_sum[192];
  __shared__ float s_cnt[192];
  __shared__ float s_red[256];
  __shared__ float s_mm[2];
  const int tid = threadIdx.x;
  const int img = blockIdx.x;
  if (tid < 192) { s_sum[tid] = 0.f; s_cnt[tid] = 0.f; }
  __syncthreads();

  const float* fab = (const float*)(ws + OFF_FAB) + (size_t)img * HH * NPAD;
  for (int p = tid; p < HH * WC; p += 256) {
    int row = p / WC;
    int col = p - row * WC;
    int r = radius[p];
    atomicAdd(&s_sum[r], fab[(size_t)row * NPAD + col]);   // ds_add_f32
    atomicAdd(&s_cnt[r], 1.0f);
  }
  __syncthreads();

  float prof = 0.f;
  if (tid < L) prof = s_sum[tid] / s_cnt[tid];
  __syncthreads();
  if (tid < 192) s_sum[tid] = (tid < L) ? prof : 0.f;
  __syncthreads();

  if (tid == 0) {
    float mn = s_sum[0];
    for (int l = 1; l < L; ++l) mn = fminf(mn, s_sum[l]);
    float mx = s_sum[0] - mn;
    for (int l = 1; l < L; ++l) mx = fmaxf(mx, s_sum[l] - mn);
    s_mm[0] = mn; s_mm[1] = mx;
  }
  __syncthreads();

  float v = 0.f;
  if (tid < L) v = fabsf((prof - s_mm[0]) / s_mm[1] - mean[tid]);
  s_red[tid] = v;
  __syncthreads();
  for (int stp = 128; stp > 0; stp >>= 1) {
    if (tid < stp) s_red[tid] += s_red[tid + stp];
    __syncthreads();
  }
  if (tid == 0) partials[img] = s_red[0];
}

__global__ void k_final(const float* __restrict__ partials, float* __restrict__ out) {
  __shared__ float s[128];
  int t = threadIdx.x;
  s[t] = partials[t];
  __syncthreads();
  for (int stp = 64; stp > 0; stp >>= 1) {
    if (t < stp) s[t] += s[t + stp];
    __syncthreads();
  }
  if (t == 0) out[0] = s[0];
}

extern "C" void kernel_launch(void* const* d_in, const int* in_sizes, int n_in,
                              void* d_out, int out_size, void* d_ws, size_t ws_size,
                              hipStream_t stream) {
  const float* data   = (const float*)d_in[0];   // [128,3,256,256] f32
  const float* mean   = (const float*)d_in[1];   // [L] f32 (L = 182)
  const int*   radius = (const int*)d_in[2];     // [256,129] i32
  char* ws = (char*)d_ws;
  const int L = in_sizes[1];

  k_init_stage1B<<<(8 * NT9 * 32 * 16 + 255) / 256, 256, 0, stream>>>(ws);
  k_init_stage2A<<<(16 * 8 * 32 * 16 + 255) / 256, 256, 0, stream>>>(ws);

  k_stage1<<<dim3(4, BIMG), 128, 0, stream>>>(data, ws);
  k_stage2<<<dim3(NT9, BIMG), 512, 0, stream>>>(ws);

  k_profile<<<BIMG, 256, 0, stream>>>(ws, radius, mean, L, (float*)(ws + OFF_PART));
  k_final<<<1, 128, 0, stream>>>((const float*)(ws + OFF_PART), (float*)d_out);
}
